// MogrifierLSTM_33225867002213
// MI455X (gfx1250) — matrix operand order repaired and checked by reading, not compiled, not run on hardware
//
#include <hip/hip_runtime.h>

// ---------------------------------------------------------------------------
// Mogrifier LSTM on MI455X (gfx1250), persistent-kernel design.
//
// Per timestep t (sequential, 256 steps):
//   P1: x  = 2*sig(h  @ Qw^T + Qb) * x_in[:,t,:]     (also stages h into act)
//   P2: hm = 2*sig(x  @ Rw^T + Rb) * h
//   P3: x  = 2*sig(hm @ Qw^T + Qb) * x
//   P4: hm = 2*sig(x  @ Rw^T + Rb) * hm
//   P5: x  = 2*sig(hm @ Qw^T + Qb) * x
//   P6: gates = [x|h] @ [W_ih|W_hh]^T + b  -> fused LSTM cell update
// Each phase = bf16 WMMA GEMM with fused epilogue; device-wide barrier after.
// 32 workgroups x 8 waves = 256 waves; each phase has exactly 256 16x16-tile
// jobs (mogrify: 4x64 tiles; gates: 4x64 jobs of 4 gate-tiles each).
// Weights (20MB bf16) are L2-resident (192MB L2); HBM traffic is negligible.
// Activations use native v_exp_f32 / v_rcp_f32; epilogues are structured as
// gather -> compute -> scatter so scattered loads clause and TRANS chains
// interleave instead of serializing per element.
// ---------------------------------------------------------------------------

typedef __attribute__((ext_vector_type(16))) __bf16 v16bf;
typedef __attribute__((ext_vector_type(8)))  __bf16 v8bf;
typedef __attribute__((ext_vector_type(8)))  float  v8f;

static constexpr int Bb = 64;
static constexpr int Tt = 256;
static constexpr int Dd = 1024;
static constexpr int Hh = 1024;
static constexpr int NBLOCKS = 32;
static constexpr int THREADS = 256;   // 8 wave32 waves per workgroup

// Native-HW activations: v_exp_f32 + v_rcp_f32 (TRANS ops, co-execute w/ VALU)
__device__ __forceinline__ float sigm(float x) {
  return __builtin_amdgcn_rcpf(1.0f + __expf(-x));
}
__device__ __forceinline__ float tanh_f(float x) {
  return 1.0f - 2.0f * __builtin_amdgcn_rcpf(__expf(2.0f * x) + 1.0f);
}

// A-fragment (16x32 bf16, MxK): lane l holds row m=l&15; lanes<16 get K 0-15,
// lanes>=16 get K 16-31 (half*16 folded into row pointer).
__device__ __forceinline__ v16bf load_a16(const __bf16* p) {
  union { v16bf v; v8bf h[2]; } u;
  u.h[0] = *(const v8bf*)(p);        // K = k0+aoff .. +7
  u.h[1] = *(const v8bf*)(p + 8);    // K = k0+8+aoff .. +7
  return u.v;
}

// Two-phase device-wide barrier (all 32 blocks resident on MI455X).
__device__ __forceinline__ void grid_barrier(int* ctrl) {
  __threadfence();
  __syncthreads();
  if (threadIdx.x == 0) {
    int* cnt = ctrl;
    int* gen = ctrl + 1;
    int g = __hip_atomic_load(gen, __ATOMIC_RELAXED, __HIP_MEMORY_SCOPE_AGENT);
    int prev = __hip_atomic_fetch_add(cnt, 1, __ATOMIC_ACQ_REL, __HIP_MEMORY_SCOPE_AGENT);
    if (prev == NBLOCKS - 1) {
      __hip_atomic_store(cnt, 0, __ATOMIC_RELAXED, __HIP_MEMORY_SCOPE_AGENT);
      __hip_atomic_fetch_add(gen, 1, __ATOMIC_ACQ_REL, __HIP_MEMORY_SCOPE_AGENT);
    } else {
      while (__hip_atomic_load(gen, __ATOMIC_ACQUIRE, __HIP_MEMORY_SCOPE_AGENT) == g) {
        __builtin_amdgcn_s_sleep(2);
      }
    }
  }
  __syncthreads();
  __threadfence();
}

// One mogrifier GEMM phase: out[64,1024] = 2*sig(A[64,1024] @ W^T + bias) * old
// Each wave owns one 16x16 output tile; K=1024 -> 32 x v_wmma_f32_16x16x32_bf16.
// HCOPY (compile-time) additionally stages carry-h into act[:,1024:].
template <bool HCOPY>
__device__ __forceinline__ void mog_gemm(
    const __bf16* __restrict__ A, int ldA,
    const __bf16* __restrict__ W,          // [1024 x 1024] row-major over K
    const float* __restrict__ bias,        // [1024]
    const float* __restrict__ oldv, size_t ld_old,
    float* __restrict__ outf,              // [64 x 1024] f32 master
    __bf16* __restrict__ outbf, int ld_outbf,
    const __bf16* __restrict__ hcopy_src,  // used iff HCOPY
    __bf16* __restrict__ hcopy_dst,        // used iff HCOPY
    int wave, int lane)
{
  const int mt = wave >> 6;                 // 0..3  (batch tile)
  const int nt = wave & 63;                 // 0..63 (output-col tile)
  const int m0 = mt * 16, n0 = nt * 16;
  const int half = lane >> 4, lm = lane & 15;
  const __bf16* Arow = A + (size_t)(m0 + lm) * ldA + half * 16;
  const __bf16* Wrow = W + (size_t)(n0 + lm) * 1024 + half * 16;  // B: K contiguous per lane
  v8f acc = {};
#pragma unroll 4
  for (int k0 = 0; k0 < 1024; k0 += 32) {
    v16bf a = load_a16(Arow + k0);
    v16bf b = *(const v16bf*)(Wrow + k0);   // 32B contiguous -> 2x b128
    acc = __builtin_amdgcn_wmma_f32_16x16x32_bf16(false, a, false, b,
                                                  (short)0, acc, false, false);
  }
  // C layout: lane holds column n0+lm; VGPR v holds row m0 + half*8 + v.
  const int col = n0 + lm;
  const int r0 = m0 + half * 8;
  const float bv = bias[col];

  // Gather phase: issue all scattered loads up front so they clause.
  float oldvals[8];
#pragma unroll
  for (int v = 0; v < 8; ++v) oldvals[v] = oldv[(size_t)(r0 + v) * ld_old + col];
  __bf16 hvals[8];
  if (HCOPY) {
#pragma unroll
    for (int v = 0; v < 8; ++v) hvals[v] = hcopy_src[(size_t)(r0 + v) * 1024 + col];
  }

  // Compute phase: 8 independent exp/rcp chains interleave.
  float nv[8];
#pragma unroll
  for (int v = 0; v < 8; ++v) nv[v] = (2.0f * sigm(acc[v] + bv)) * oldvals[v];

  // Scatter phase.
#pragma unroll
  for (int v = 0; v < 8; ++v) {
    outf[(size_t)(r0 + v) * 1024 + col] = nv[v];
    outbf[(size_t)(r0 + v) * ld_outbf + col] = (__bf16)nv[v];
  }
  if (HCOPY) {
#pragma unroll
    for (int v = 0; v < 8; ++v)
      hcopy_dst[(size_t)(r0 + v) * 2048 + 1024 + col] = hvals[v];
  }
}

// Gate GEMM + fused LSTM cell update. Each wave computes the 4 gate tiles
// (i,f,g,o) for one (batch-tile, h-col-tile): A fragment shared, 4 WMMAs/k-step.
__device__ __forceinline__ void lstm_gates(
    const __bf16* __restrict__ act,    // [64 x 2048] = [x_bf | h_bf]
    const __bf16* __restrict__ Wcat,   // [4096 x 2048] = [W_ih | W_hh] rows
    const float* __restrict__ bih, const float* __restrict__ bhh,
    float* __restrict__ hf, float* __restrict__ cf, __bf16* __restrict__ hbf,
    float* __restrict__ out, int t,
    float* __restrict__ hout, float* __restrict__ cout,
    int wave, int lane)
{
  const int mt = wave >> 6, nt = wave & 63;
  const int m0 = mt * 16, n0 = nt * 16;
  const int half = lane >> 4, lm = lane & 15;
  const __bf16* Arow = act + (size_t)(m0 + lm) * 2048 + half * 16;
  const __bf16* W0 = Wcat + (size_t)(n0 + lm) * 2048 + half * 16;
  const __bf16* W1 = W0 + (size_t)1024 * 2048;
  const __bf16* W2 = W0 + (size_t)2048 * 2048;
  const __bf16* W3 = W0 + (size_t)3072 * 2048;
  v8f ai = {}, af = {}, ag = {}, ao = {};
#pragma unroll 2
  for (int k0 = 0; k0 < 2048; k0 += 32) {
    v16bf a = load_a16(Arow + k0);
    v16bf b0 = *(const v16bf*)(W0 + k0);
    ai = __builtin_amdgcn_wmma_f32_16x16x32_bf16(false, a, false, b0, (short)0, ai, false, false);
    v16bf b1 = *(const v16bf*)(W1 + k0);
    af = __builtin_amdgcn_wmma_f32_16x16x32_bf16(false, a, false, b1, (short)0, af, false, false);
    v16bf b2 = *(const v16bf*)(W2 + k0);
    ag = __builtin_amdgcn_wmma_f32_16x16x32_bf16(false, a, false, b2, (short)0, ag, false, false);
    v16bf b3 = *(const v16bf*)(W3 + k0);
    ao = __builtin_amdgcn_wmma_f32_16x16x32_bf16(false, a, false, b3, (short)0, ao, false, false);
  }
  const int col = n0 + lm;
  const int r0 = m0 + half * 8;
  const float bi  = bih[col]        + bhh[col];
  const float bfv = bih[1024 + col] + bhh[1024 + col];
  const float bg  = bih[2048 + col] + bhh[2048 + col];
  const float bo  = bih[3072 + col] + bhh[3072 + col];

  // Gather phase: c values for all 8 rows in flight together.
  float cold[8];
#pragma unroll
  for (int v = 0; v < 8; ++v) cold[v] = cf[(size_t)(r0 + v) * 1024 + col];

  // Compute phase: 32 independent transcendental chains interleave.
  float cn[8], hn[8];
#pragma unroll
  for (int v = 0; v < 8; ++v) {
    const float ig = sigm(ai[v] + bi);
    const float fg = sigm(af[v] + bfv);
    const float gg = tanh_f(ag[v] + bg);
    const float og = sigm(ao[v] + bo);
    cn[v] = fg * cold[v] + ig * gg;
    hn[v] = og * tanh_f(cn[v]);
  }

  // Scatter phase.
#pragma unroll
  for (int v = 0; v < 8; ++v) {
    const size_t idx = (size_t)(r0 + v) * 1024 + col;
    cf[idx] = cn[v];
    hf[idx] = hn[v];
    hbf[idx] = (__bf16)hn[v];                          // next step's P1 input
    out[((size_t)(r0 + v) * Tt + t) * Hh + col] = hn[v];
  }
  if (t == Tt - 1) {
#pragma unroll
    for (int v = 0; v < 8; ++v) {
      const size_t idx = (size_t)(r0 + v) * 1024 + col;
      hout[idx] = hn[v];
      cout[idx] = cn[v];
    }
  }
}

__global__ void __launch_bounds__(THREADS)
mogrifier_lstm_persistent(
    const float* __restrict__ xin,
    const float* __restrict__ Qb, const float* __restrict__ Rb,
    const float* __restrict__ bih, const float* __restrict__ bhh,
    const __bf16* __restrict__ Qw, const __bf16* __restrict__ Rw,
    const __bf16* __restrict__ Wcat,
    __bf16* act, __bf16* hbf, __bf16* hmbf,
    float* xf, float* hmf, float* hf, float* cf,
    float* out, float* hout, float* cout,
    int* ctrl)
{
  const int wave = blockIdx.x * (THREADS / 32) + (threadIdx.x >> 5);
  const int lane = threadIdx.x & 31;
  for (int t = 0; t < Tt; ++t) {
    // P1: x = 2*sig(h @ Qw^T + Qb) * x_in[:,t,:]; stage h_bf into act[:,1024:]
    mog_gemm<true>(hbf, 1024, Qw, Qb, xin + (size_t)t * Dd, (size_t)Tt * Dd,
                   xf, act, 2048, hbf, act, wave, lane);
    grid_barrier(ctrl);
    // P2: hm = 2*sig(x @ Rw^T + Rb) * h
    mog_gemm<false>(act, 2048, Rw, Rb, hf, 1024, hmf, hmbf, 1024,
                    nullptr, nullptr, wave, lane);
    grid_barrier(ctrl);
    // P3: x = 2*sig(hm @ Qw^T + Qb) * x
    mog_gemm<false>(hmbf, 1024, Qw, Qb, xf, 1024, xf, act, 2048,
                    nullptr, nullptr, wave, lane);
    grid_barrier(ctrl);
    // P4: hm = 2*sig(x @ Rw^T + Rb) * hm
    mog_gemm<false>(act, 2048, Rw, Rb, hmf, 1024, hmf, hmbf, 1024,
                    nullptr, nullptr, wave, lane);
    grid_barrier(ctrl);
    // P5: x = 2*sig(hm @ Qw^T + Qb) * x
    mog_gemm<false>(hmbf, 1024, Qw, Qb, xf, 1024, xf, act, 2048,
                    nullptr, nullptr, wave, lane);
    grid_barrier(ctrl);
    // P6: gates + LSTM cell update (i,f,g,o fused, no gates buffer)
    lstm_gates(act, Wcat, bih, bhh, hf, cf, hbf, out, t, hout, cout, wave, lane);
    grid_barrier(ctrl);
  }
}

// Convert weights to bf16 once per launch; build Wcat = [W_ih | W_hh] rows.
__global__ void prep_weights(const float* __restrict__ Qw_f,
                             const float* __restrict__ Rw_f,
                             const float* __restrict__ Wih,
                             const float* __restrict__ Whh,
                             __bf16* Qw, __bf16* Rw, __bf16* Wcat)
{
  const size_t i = (size_t)blockIdx.x * blockDim.x + threadIdx.x;
  const size_t stride = (size_t)gridDim.x * blockDim.x;
  const size_t NQ = (size_t)Dd * Hh;
  for (size_t j = i; j < NQ; j += stride) Qw[j] = (__bf16)Qw_f[j];
  for (size_t j = i; j < NQ; j += stride) Rw[j] = (__bf16)Rw_f[j];
  const size_t NC = (size_t)4 * Hh * 2048;
  for (size_t j = i; j < NC; j += stride) {
    const size_t o = j >> 11;          // row (0..4095)
    const size_t k = j & 2047;         // col (0..2047)
    const float v = (k < 1024) ? Wih[o * 1024 + k] : Whh[o * 1024 + (k - 1024)];
    Wcat[j] = (__bf16)v;
  }
}

// Deterministic per-launch init: h0 = c0 = 0, barrier counters = 0.
__global__ void init_state(float* hf, float* cf, __bf16* hbf, int* ctrl)
{
  const size_t i = (size_t)blockIdx.x * blockDim.x + threadIdx.x;
  const size_t stride = (size_t)gridDim.x * blockDim.x;
  const size_t N = (size_t)Bb * Hh;
  for (size_t j = i; j < N; j += stride) {
    hf[j] = 0.0f;
    cf[j] = 0.0f;
    hbf[j] = (__bf16)0.0f;
  }
  if (i < 2) ctrl[i] = 0;
}

extern "C" void kernel_launch(void* const* d_in, const int* in_sizes, int n_in,
                              void* d_out, int out_size, void* d_ws, size_t ws_size,
                              hipStream_t stream) {
  (void)in_sizes; (void)n_in; (void)out_size; (void)ws_size;
  const float* x    = (const float*)d_in[0];
  const float* Qw_f = (const float*)d_in[1];
  const float* Qb   = (const float*)d_in[2];
  const float* Rw_f = (const float*)d_in[3];
  const float* Rb   = (const float*)d_in[4];
  const float* Wih  = (const float*)d_in[5];
  const float* Whh  = (const float*)d_in[6];
  const float* bih  = (const float*)d_in[7];
  const float* bhh  = (const float*)d_in[8];

  char* ws = (char*)d_ws;
  size_t off = 0;
  auto take = [&](size_t bytes) -> void* {
    void* p = ws + off;
    off += (bytes + 255) & ~(size_t)255;
    return p;
  };
  int*    ctrl = (int*)take(256);
  __bf16* Qw   = (__bf16*)take((size_t)Dd * Hh * 2);        //  2 MiB
  __bf16* Rw   = (__bf16*)take((size_t)Hh * Dd * 2);        //  2 MiB
  __bf16* Wcat = (__bf16*)take((size_t)4 * Hh * 2048 * 2);  // 16 MiB
  __bf16* act  = (__bf16*)take((size_t)Bb * 2048 * 2);      // [x_bf | h_bf]
  __bf16* hbf  = (__bf16*)take((size_t)Bb * Hh * 2);
  __bf16* hmbf = (__bf16*)take((size_t)Bb * Hh * 2);
  float*  xf   = (float*)take((size_t)Bb * Dd * 4);
  float*  hmf  = (float*)take((size_t)Bb * Hh * 4);
  float*  hf   = (float*)take((size_t)Bb * Hh * 4);
  float*  cf   = (float*)take((size_t)Bb * Hh * 4);

  float* out  = (float*)d_out;                       // [B,T,H]
  float* hout = out + (size_t)Bb * Tt * Hh;          // [1,B,H]
  float* cout = hout + (size_t)Bb * Hh;              // [1,B,H]

  prep_weights<<<2048, 256, 0, stream>>>(Qw_f, Rw_f, Wih, Whh, Qw, Rw, Wcat);
  init_state<<<64, 256, 0, stream>>>(hf, cf, hbf, ctrl);
  mogrifier_lstm_persistent<<<NBLOCKS, THREADS, 0, stream>>>(
      x, Qb, Rb, bih, bhh, Qw, Rw, Wcat, act, hbf, hmbf, xf, hmf, hf, cf,
      out, hout, cout, ctrl);
}
